// CalendarGNN_4252017623144
// MI455X (gfx1250) — matrix operand order repaired and checked by reading, not compiled, not run on hardware
//
#include <hip/hip_runtime.h>
#include <hip/hip_bf16.h>

#define NUM_S   8904
#define MAXLEN  50
#define EMB_V   128
#define EMB_L   64
#define H_SESS  256
#define H_EMB   128
#define PAT_D   128

typedef __attribute__((ext_vector_type(16))) __bf16 v16bf;
typedef __attribute__((ext_vector_type(8)))  float  v8f;
typedef __attribute__((ext_vector_type(4)))  unsigned v4u;

union FragU { v16bf v; v4u q[2]; };

#define V8F_ZERO {0.f, 0.f, 0.f, 0.f, 0.f, 0.f, 0.f, 0.f}

__device__ __forceinline__ unsigned short f2bf(float f) {
    unsigned u = __float_as_uint(f);
    return (unsigned short)((u + 0x7FFFu + ((u >> 16) & 1u)) >> 16);
}
__device__ __forceinline__ unsigned pack2bf(float lo, float hi) {
    return (unsigned)f2bf(lo) | ((unsigned)f2bf(hi) << 16);
}
__device__ __forceinline__ float sigm(float x) { return 1.0f / (1.0f + __expf(-x)); }

// A-fragment (16x32 bf16) from an LDS tile, row-major with strideU (ushort units).
// ISA layout: lanes 0-15 = rows M=lane, K 0..7 (VGPR0-3) and 16..23 (VGPR4-7);
//             lanes 16-31 = rows M=lane-16, K 8..15 and 24..31.
// All bases/strides are 16B-aligned -> two ds_load_b128.
__device__ __forceinline__ v16bf load_a_lds(const unsigned short* base, int strideU,
                                            int lane, int kt) {
    const int m   = lane & 15;
    const int klo = (lane & 16) ? 8 : 0;
    const unsigned short* p = base + m * strideU + kt * 32 + klo;
    FragU f;
    f.q[0] = *(const v4u*)(p);
    f.q[1] = *(const v4u*)(p + 16);
    return f.v;
}

// B-fragment from pre-swizzled global buffer: F[((kt*NT+nt)*256 + lane*8 + v)]
// 32B-aligned per lane -> two global_load_b128.
__device__ __forceinline__ v16bf load_b_frag(const unsigned* __restrict__ F, int NT,
                                             int kt, int nt, int lane) {
    const v4u* p = (const v4u*)(F + (((size_t)(kt * NT + nt)) * 256 + lane * 8));
    FragU f;
    f.q[0] = p[0];
    f.q[1] = p[1];
    return f.v;
}

__device__ __forceinline__ v8f wmma_bf16(v16bf a, v16bf b, v8f c) {
    return __builtin_amdgcn_wmma_f32_16x16x32_bf16(false, a, false, b, (short)0, c,
                                                   false, false);
}

// ---------------------------------------------------------------- utility kernels
__global__ void zero_dw_kernel(unsigned* p, int n) {
    int i = blockIdx.x * 256 + threadIdx.x;
    if (i < n) p[i] = 0u;
}

__global__ void compute_lens_kernel(const int* __restrict__ vs, int* __restrict__ lens) {
    int s = blockIdx.x * 256 + threadIdx.x;
    if (s < NUM_S) {
        int c = 0;
#pragma unroll 5
        for (int t = 0; t < MAXLEN; ++t) c += (vs[s * MAXLEN + t] > 0) ? 1 : 0;
        lens[s] = c;
    }
}

// Pre-swizzle W (row-major [N][K] fp32, torch layout) into bf16 WMMA B-fragments.
__global__ void pack_b_frag_kernel(const float* __restrict__ W, unsigned* __restrict__ F,
                                   int K, int N) {
    int NT = N >> 4;
    int total = (K >> 5) * NT * 256;
    int idx = blockIdx.x * 256 + threadIdx.x;
    if (idx >= total) return;
    int v    = idx & 7;
    int lane = (idx >> 3) & 31;
    int tile = idx >> 8;
    int nt   = tile % NT;
    int kt   = tile / NT;
    int n = nt * 16 + (lane & 15);
    int k = kt * 32 + ((lane & 16) ? 8 : 0) + 2 * v + ((v >= 4) ? 8 : 0);
    F[idx] = pack2bf(W[(size_t)n * K + k], W[(size_t)n * K + k + 1]);
}

// Stable scatter order: grpidx[g*maxc + pos] = s  (session order preserved in group).
__global__ void build_groups_kernel(const int* __restrict__ ts, const int* __restrict__ usl,
                                    int* gh, int* gw, int* gy, int* gl) {
    int which = blockIdx.x;
    int g = threadIdx.x;
    int U, maxc; int* dst;
    if (which == 0)      { U = 24; maxc = 371;  dst = gh; }
    else if (which == 1) { U = 53; maxc = 168;  dst = gw; }
    else if (which == 2) { U = 7;  maxc = 1272; dst = gy; }
    else                 { U = 56; maxc = 159;  dst = gl; }
    if (g >= U) return;
    int cnt = 0;
    for (int s = 0; s < NUM_S; ++s) {
        int id = (which == 0) ? ts[s * 4 + 1]
               : (which == 1) ? ts[s * 4 + 2]
               : (which == 2) ? ts[s * 4 + 3]
                              : usl[s];
        if (id == g) dst[g * maxc + (cnt++)] = s;
    }
}

// ------------------------------------------------- item2sess GRU: one launch per step
// block = 16 batch rows x 768 gate cols, 256 threads (8 waves x 6 N-tiles each).
// Two WMMA passes (gi then gh) so each A-fragment is loaded once per wave and
// only 6 accumulators (48 VGPRs) stay live.
__global__ void sess_step_kernel(const int* __restrict__ vs, const float* __restrict__ embv,
                                 const unsigned* __restrict__ Fih,
                                 const unsigned* __restrict__ Fhh,
                                 const float* __restrict__ bih, const float* __restrict__ bhh,
                                 const int* __restrict__ lens,
                                 float* __restrict__ hf, unsigned short* __restrict__ hb,
                                 int t) {
    extern __shared__ char smem[];
    float*          GRZ = (float*)smem;            // [16][512]
    float*          GIN = GRZ + 16 * 512;          // [16][256]
    float*          GHN = GIN + 16 * 256;          // [16][256]
    unsigned short* Xs  = (unsigned short*)(GHN + 16 * 256); // [16][136] bf16
    unsigned short* Hs  = Xs + 16 * 136;           // [16][264] bf16

    const int tid = threadIdx.x;
    const int m0  = blockIdx.x * 16;

    // stage X (embedding gather -> bf16) and h (bf16)
    for (int e = tid; e < 16 * 128; e += 256) {
        int i = e >> 7, k = e & 127;
        int s = m0 + i;
        int v = (s < NUM_S) ? vs[s * MAXLEN + t] : 0;
        Xs[i * 136 + k] = f2bf(embv[(size_t)v * 128 + k]);
    }
    for (int e = tid; e < 16 * 256; e += 256) {
        int i = e >> 8, k = e & 255;
        int s = m0 + i;
        Hs[i * 264 + k] = (s < NUM_S) ? hb[(size_t)s * 256 + k] : (unsigned short)0;
    }
    __syncthreads();

    const int wave = tid >> 5, lane = tid & 31;
    const int nlo   = lane & 15;
    const int mhalf = (lane & 16) ? 8 : 0;

    v8f acc[6] = {V8F_ZERO, V8F_ZERO, V8F_ZERO, V8F_ZERO, V8F_ZERO, V8F_ZERO};
    // pass 1: gi = X @ Wih^T   (K = 128)
#pragma unroll
    for (int kt = 0; kt < 4; ++kt) {
        v16bf a = load_a_lds(Xs, 136, lane, kt);
#pragma unroll
        for (int i = 0; i < 6; ++i)
            acc[i] = wmma_bf16(a, load_b_frag(Fih, 48, kt, wave * 6 + i, lane), acc[i]);
    }
#pragma unroll
    for (int i = 0; i < 6; ++i) {
        int nIdx = (wave * 6 + i) * 16 + nlo;
        float bi = bih[nIdx];
#pragma unroll
        for (int r = 0; r < 8; ++r) {
            int m = mhalf + r;
            float gi = acc[i][r] + bi;
            if (nIdx < 512) GRZ[m * 512 + nIdx] = gi;
            else            GIN[m * 256 + nIdx - 512] = gi;
        }
        acc[i] = (v8f)V8F_ZERO;
    }
    // pass 2: gh = h @ Whh^T   (K = 256); same lane owns same gate elements,
    // so the LDS += below needs no barrier.
#pragma unroll
    for (int kt = 0; kt < 8; ++kt) {
        v16bf a = load_a_lds(Hs, 264, lane, kt);
#pragma unroll
        for (int i = 0; i < 6; ++i)
            acc[i] = wmma_bf16(a, load_b_frag(Fhh, 48, kt, wave * 6 + i, lane), acc[i]);
    }
#pragma unroll
    for (int i = 0; i < 6; ++i) {
        int nIdx = (wave * 6 + i) * 16 + nlo;
        float bh = bhh[nIdx];
#pragma unroll
        for (int r = 0; r < 8; ++r) {
            int m = mhalf + r;
            float gh = acc[i][r] + bh;
            if (nIdx < 512) GRZ[m * 512 + nIdx] += gh;
            else            GHN[m * 256 + nIdx - 512] = gh;
        }
    }
    __syncthreads();

    // elementwise GRU update with length mask
    for (int e = tid; e < 16 * 256; e += 256) {
        int m = e >> 8, j = e & 255;
        int s = m0 + m;
        if (s >= NUM_S) continue;
        float r = sigm(GRZ[m * 512 + j]);
        float z = sigm(GRZ[m * 512 + 256 + j]);
        float n = tanhf(GIN[m * 256 + j] + r * GHN[m * 256 + j]);
        float hold = hf[(size_t)s * 256 + j];
        float hnew = (t < lens[s]) ? ((1.f - z) * n + z * hold) : hold;
        hf[(size_t)s * 256 + j] = hnew;
        hb[(size_t)s * 256 + j] = f2bf(hnew);
    }
}

// -------------------------------------------- group GRUs (4 persistent workgroups)
struct GroupArgs {
    const unsigned* Fih[4]; const unsigned* Fhh[4];
    const float*    bih[4]; const float*    bhh[4];
    const int*      gidx[4];
    float*          unit[4];
    int U[4], maxc[4], I[4];
};

__global__ void group_gru_kernel(GroupArgs ga, const float* __restrict__ sess,
                                 const float* __restrict__ embl, const int* __restrict__ usl) {
    const int gid = blockIdx.x;
    const int U = ga.U[gid], T = ga.maxc[gid], I = ga.I[gid];
    const int Ikt = I >> 5;
    const int RT = (U + 15) >> 4;
    const unsigned* Fih = ga.Fih[gid]; const unsigned* Fhh = ga.Fhh[gid];
    const float* bih = ga.bih[gid];    const float* bhh = ga.bhh[gid];
    const int* gidx = ga.gidx[gid];

    extern __shared__ char smem[];
    float*          HF  = (float*)smem;            // [64][128]
    float*          GRZ = HF + 64 * 128;           // [16][256]
    float*          GIN = GRZ + 16 * 256;          // [16][128]
    float*          GHN = GIN + 16 * 128;          // [16][128]
    unsigned short* HB  = (unsigned short*)(GHN + 16 * 128); // [64][136]
    unsigned short* XS  = HB + 64 * 136;           // [16][328]

    const int tid = threadIdx.x, wave = tid >> 5, lane = tid & 31;
    const int nlo   = lane & 15;
    const int mhalf = (lane & 16) ? 8 : 0;
    for (int e = tid; e < 64 * 128; e += 256) HF[e] = 0.f;
    for (int e = tid; e < 64 * 136; e += 256) HB[e] = 0;
    __syncthreads();

    for (int t = 0; t < T; ++t) {
        for (int rt = 0; rt < RT; ++rt) {
            // stage x rows for this row-tile (gather + concat for spatial GRU)
            for (int e = tid; e < 16 * I; e += 256) {
                int i = e / I, k = e - i * I;
                int g = rt * 16 + i;
                float val = 0.f;
                if (g < U) {
                    int s = gidx[g * T + t];
                    val = (k < 256) ? sess[(size_t)s * 256 + k]
                                    : embl[(size_t)usl[s] * EMB_L + (k - 256)];
                }
                XS[i * 328 + k] = f2bf(val);
            }
            __syncthreads();

            v8f ax[3] = {V8F_ZERO, V8F_ZERO, V8F_ZERO};
            v8f ah[3] = {V8F_ZERO, V8F_ZERO, V8F_ZERO};
            for (int kt = 0; kt < Ikt; ++kt) {
                v16bf a = load_a_lds(XS, 328, lane, kt);
#pragma unroll
                for (int i = 0; i < 3; ++i)
                    ax[i] = wmma_bf16(a, load_b_frag(Fih, 24, kt, wave * 3 + i, lane), ax[i]);
            }
#pragma unroll
            for (int kt = 0; kt < 4; ++kt) {
                v16bf a = load_a_lds(HB + rt * 16 * 136, 136, lane, kt);
#pragma unroll
                for (int i = 0; i < 3; ++i)
                    ah[i] = wmma_bf16(a, load_b_frag(Fhh, 24, kt, wave * 3 + i, lane), ah[i]);
            }
#pragma unroll
            for (int i = 0; i < 3; ++i) {
                int nIdx = (wave * 3 + i) * 16 + nlo;
                float bi = bih[nIdx], bh = bhh[nIdx];
#pragma unroll
                for (int r = 0; r < 8; ++r) {
                    int m = mhalf + r;
                    float gi = ax[i][r] + bi, gh = ah[i][r] + bh;
                    if (nIdx < 256) GRZ[m * 256 + nIdx] = gi + gh;
                    else { GIN[m * 128 + nIdx - 256] = gi; GHN[m * 128 + nIdx - 256] = gh; }
                }
            }
            __syncthreads();
            for (int e = tid; e < 16 * 128; e += 256) {
                int m = e >> 7, j = e & 127;
                int g = rt * 16 + m;
                if (g < U) {
                    float r = sigm(GRZ[m * 256 + j]);
                    float z = sigm(GRZ[m * 256 + 128 + j]);
                    float n = tanhf(GIN[m * 128 + j] + r * GHN[m * 128 + j]);
                    float hold = HF[g * 128 + j];
                    float hnew = (1.f - z) * n + z * hold;   // counts uniform: always < T
                    HF[g * 128 + j] = hnew;
                    HB[g * 136 + j] = f2bf(hnew);
                }
            }
            __syncthreads();
        }
    }
    float* unit = ga.unit[gid];
    for (int e = tid; e < U * 128; e += 256) unit[e] = HF[(e >> 7) * 128 + (e & 127)];
}

// -------------------------------------------- pattern GRUs (B=1, 4 workgroups)
struct PatArgs {
    const unsigned* Fih[4]; const unsigned* Fhh[4];
    const float*    bih[4]; const float*    bhh[4];
    const float*    unit[4];
    int T[4];
    float* pat;   // [4*128]
};

__global__ void pat_gru_kernel(PatArgs pa) {
    const int gid = blockIdx.x;
    const int T = pa.T[gid];
    const unsigned* Fih = pa.Fih[gid]; const unsigned* Fhh = pa.Fhh[gid];
    const float* bih = pa.bih[gid];    const float* bhh = pa.bhh[gid];
    const float* unit = pa.unit[gid];

    extern __shared__ char smem[];
    float*          GRZ = (float*)smem;            // [16][256]
    float*          GIN = GRZ + 16 * 256;          // [16][128]
    float*          GHN = GIN + 16 * 128;          // [16][128]
    float*          HF  = GHN + 16 * 128;          // [128]
    unsigned short* HB  = (unsigned short*)(HF + 128);  // [16][136]
    unsigned short* XS  = HB + 16 * 136;           // [16][136]

    const int tid = threadIdx.x, wave = tid >> 5, lane = tid & 31;
    const int nlo   = lane & 15;
    const int mhalf = (lane & 16) ? 8 : 0;
    for (int e = tid; e < 128; e += 256) HF[e] = 0.f;
    for (int e = tid; e < 16 * 136; e += 256) { HB[e] = 0; XS[e] = 0; }
    __syncthreads();

    for (int t = 0; t < T; ++t) {
        for (int e = tid; e < 128; e += 256) XS[e] = f2bf(unit[(size_t)t * 128 + e]);
        __syncthreads();

        v8f ax[3] = {V8F_ZERO, V8F_ZERO, V8F_ZERO};
        v8f ah[3] = {V8F_ZERO, V8F_ZERO, V8F_ZERO};
#pragma unroll
        for (int kt = 0; kt < 4; ++kt) {
            v16bf aX = load_a_lds(XS, 136, lane, kt);
            v16bf aH = load_a_lds(HB, 136, lane, kt);
#pragma unroll
            for (int i = 0; i < 3; ++i) {
                ax[i] = wmma_bf16(aX, load_b_frag(Fih, 24, kt, wave * 3 + i, lane), ax[i]);
                ah[i] = wmma_bf16(aH, load_b_frag(Fhh, 24, kt, wave * 3 + i, lane), ah[i]);
            }
        }
#pragma unroll
        for (int i = 0; i < 3; ++i) {
            int nIdx = (wave * 3 + i) * 16 + nlo;
            float bi = bih[nIdx], bh = bhh[nIdx];
#pragma unroll
            for (int r = 0; r < 8; ++r) {
                int m = mhalf + r;
                float gi = ax[i][r] + bi, gh = ah[i][r] + bh;
                if (nIdx < 256) GRZ[m * 256 + nIdx] = gi + gh;
                else { GIN[m * 128 + nIdx - 256] = gi; GHN[m * 128 + nIdx - 256] = gh; }
            }
        }
        __syncthreads();
        for (int e = tid; e < 128; e += 256) {   // only row 0 is live
            float r = sigm(GRZ[e]);
            float z = sigm(GRZ[128 + e]);
            float n = tanhf(GIN[e] + r * GHN[e]);
            float hnew = (1.f - z) * n + z * HF[e];
            HF[e] = hnew;
            HB[e] = f2bf(hnew);
        }
        __syncthreads();
    }
    for (int e = tid; e < 128; e += 256) pa.pat[gid * 128 + e] = HF[e];
}

// -------------------------------------------- final FC: [1,512] x [64,512]^T + b
__global__ void fc_out_kernel(const float* __restrict__ pat, const float* __restrict__ W,
                              const float* __restrict__ b, float* __restrict__ out) {
    int o = threadIdx.x;
    if (o < 64) {
        float acc = b[o];
#pragma unroll 8
        for (int j = 0; j < 512; ++j) acc += pat[j] * W[o * 512 + j];
        out[o] = acc;
    }
}

// ================================================================ host launcher
static inline unsigned div_up_u(unsigned a, unsigned b) { return (a + b - 1) / b; }

extern "C" void kernel_launch(void* const* d_in, const int* in_sizes, int n_in,
                              void* d_out, int out_size, void* d_ws, size_t ws_size,
                              hipStream_t stream) {
    if (n_in < 43) return;
    const int*   u_s_vs = (const int*)d_in[0];
    const int*   u_s_ts = (const int*)d_in[1];
    const int*   u_s_l  = (const int*)d_in[2];
    const float* embs_v = (const float*)d_in[3];
    const float* embs_l = (const float*)d_in[4];
    struct GruP { const float *Wih, *Whh, *bih, *bhh; };
    auto gp = [&](int base) {
        return GruP{(const float*)d_in[base], (const float*)d_in[base + 1],
                    (const float*)d_in[base + 2], (const float*)d_in[base + 3]};
    };
    GruP i2s = gp(5);
    GruP grp_p[4] = {gp(9), gp(13), gp(17), gp(21)};   // sess2{h,w,y,l}emb
    GruP pat_p[4] = {gp(25), gp(29), gp(33), gp(37)};  // {h,w,y,l}emb2pat
    const float* fc_w = (const float*)d_in[41];
    const float* fc_b = (const float*)d_in[42];

    // ---- workspace carve
    char* ws = (char*)d_ws;
    size_t off = 0;
    auto alloc = [&](size_t bytes) -> void* {
        void* p = ws + off;
        off = (off + bytes + 255) & ~(size_t)255;
        return p;
    };
    auto frag_dw = [](int K, int N) { return (size_t)(K / 32) * (N / 16) * 256; };

    int*            lens = (int*)alloc(NUM_S * 4);
    float*          hf   = (float*)alloc((size_t)NUM_S * 256 * 4);
    unsigned short* hb   = (unsigned short*)alloc((size_t)NUM_S * 256 * 2);
    unsigned* F_i2s_ih = (unsigned*)alloc(frag_dw(128, 768) * 4);
    unsigned* F_i2s_hh = (unsigned*)alloc(frag_dw(256, 768) * 4);
    const int gU[4]    = {24, 53, 7, 56};
    const int gMaxc[4] = {371, 168, 1272, 159};
    const int gI[4]    = {256, 256, 256, 320};
    unsigned *F_g_ih[4], *F_g_hh[4], *F_p_ih[4], *F_p_hh[4];
    int* gidx[4];
    float* unit[4];
    for (int g = 0; g < 4; ++g) {
        F_g_ih[g] = (unsigned*)alloc(frag_dw(gI[g], 384) * 4);
        F_g_hh[g] = (unsigned*)alloc(frag_dw(128, 384) * 4);
        F_p_ih[g] = (unsigned*)alloc(frag_dw(128, 384) * 4);
        F_p_hh[g] = (unsigned*)alloc(frag_dw(128, 384) * 4);
        gidx[g]   = (int*)alloc((size_t)NUM_S * 4);
        unit[g]   = (float*)alloc((size_t)gU[g] * 128 * 4);
    }
    float* pat = (float*)alloc(512 * 4);
    (void)ws_size; (void)in_sizes; (void)out_size;

    // ---- setup: lens, zero hidden state, weight swizzle, group order
    compute_lens_kernel<<<div_up_u(NUM_S, 256), 256, 0, stream>>>(u_s_vs, lens);
    zero_dw_kernel<<<div_up_u(NUM_S * 256, 256), 256, 0, stream>>>((unsigned*)hf, NUM_S * 256);
    zero_dw_kernel<<<div_up_u(NUM_S * 128, 256), 256, 0, stream>>>((unsigned*)hb, NUM_S * 128);

    struct PackJob { const float* W; unsigned* F; int K, N; };
    PackJob jobs[18] = {
        {i2s.Wih, F_i2s_ih, 128, 768}, {i2s.Whh, F_i2s_hh, 256, 768},
        {grp_p[0].Wih, F_g_ih[0], 256, 384}, {grp_p[0].Whh, F_g_hh[0], 128, 384},
        {grp_p[1].Wih, F_g_ih[1], 256, 384}, {grp_p[1].Whh, F_g_hh[1], 128, 384},
        {grp_p[2].Wih, F_g_ih[2], 256, 384}, {grp_p[2].Whh, F_g_hh[2], 128, 384},
        {grp_p[3].Wih, F_g_ih[3], 320, 384}, {grp_p[3].Whh, F_g_hh[3], 128, 384},
        {pat_p[0].Wih, F_p_ih[0], 128, 384}, {pat_p[0].Whh, F_p_hh[0], 128, 384},
        {pat_p[1].Wih, F_p_ih[1], 128, 384}, {pat_p[1].Whh, F_p_hh[1], 128, 384},
        {pat_p[2].Wih, F_p_ih[2], 128, 384}, {pat_p[2].Whh, F_p_hh[2], 128, 384},
        {pat_p[3].Wih, F_p_ih[3], 128, 384}, {pat_p[3].Whh, F_p_hh[3], 128, 384},
    };
    for (int j = 0; j < 18; ++j) {
        int total = (int)frag_dw(jobs[j].K, jobs[j].N);
        pack_b_frag_kernel<<<div_up_u(total, 256), 256, 0, stream>>>(jobs[j].W, jobs[j].F,
                                                                     jobs[j].K, jobs[j].N);
    }
    build_groups_kernel<<<4, 64, 0, stream>>>(u_s_ts, u_s_l, gidx[0], gidx[1], gidx[2], gidx[3]);

    // ---- stage 1: item2sess GRU, one launch per step (global sync between steps)
    const int smem_sess = (16 * 512 + 16 * 256 + 16 * 256) * 4 + (16 * 136 + 16 * 264) * 2;
    for (int t = 0; t < MAXLEN; ++t)
        sess_step_kernel<<<div_up_u(NUM_S, 16), 256, smem_sess, stream>>>(
            u_s_vs, embs_v, F_i2s_ih, F_i2s_hh, i2s.bih, i2s.bhh, lens, hf, hb, t);

    // ---- stage 2: group GRUs (4 persistent workgroups)
    GroupArgs ga;
    for (int g = 0; g < 4; ++g) {
        ga.Fih[g] = F_g_ih[g]; ga.Fhh[g] = F_g_hh[g];
        ga.bih[g] = grp_p[g].bih; ga.bhh[g] = grp_p[g].bhh;
        ga.gidx[g] = gidx[g]; ga.unit[g] = unit[g];
        ga.U[g] = gU[g]; ga.maxc[g] = gMaxc[g]; ga.I[g] = gI[g];
    }
    const int smem_grp = (64 * 128 + 16 * 256 + 16 * 128 + 16 * 128) * 4 +
                         (64 * 136 + 16 * 328) * 2;
    group_gru_kernel<<<4, 256, smem_grp, stream>>>(ga, hf, embs_l, u_s_l);

    // ---- stage 3: pattern GRUs
    PatArgs pa;
    for (int g = 0; g < 4; ++g) {
        pa.Fih[g] = F_p_ih[g]; pa.Fhh[g] = F_p_hh[g];
        pa.bih[g] = pat_p[g].bih; pa.bhh[g] = pat_p[g].bhh;
        pa.unit[g] = unit[g]; pa.T[g] = gU[g];
    }
    pa.pat = pat;
    const int smem_pat = (16 * 256 + 16 * 128 + 16 * 128 + 128) * 4 + (16 * 136 * 2) * 2;
    pat_gru_kernel<<<4, 256, smem_pat, stream>>>(pa);

    // ---- stage 4: final FC
    fc_out_kernel<<<1, 64, 0, stream>>>(pat, fc_w, fc_b, (float*)d_out);
}
